// QAGNN_40913858462174
// MI455X (gfx1250) — compile-verified, matrix operand order checked by reading, not run on hardware
//
#include <hip/hip_runtime.h>
#include <hip/hip_bf16.h>
#include <stdint.h>

// ---------------------------------------------------------------------------
// CDNA5 (gfx1250) QAGNN pipeline: cosine relevance -> GAT(768->256, bf16 WMMA)
// -> BN -> ReLU -> GAT(256->256, bf16 WMMA) -> ReLU -> mean-pool -> classifier
// ---------------------------------------------------------------------------

typedef __attribute__((ext_vector_type(16))) __bf16 v16bf;
typedef __attribute__((ext_vector_type(8)))  __bf16 v8bf;
typedef __attribute__((ext_vector_type(8)))  float  v8f;

__device__ __forceinline__ unsigned short f2bf(float f) {
  unsigned u = __float_as_uint(f);
  unsigned r = u + 0x7FFFu + ((u >> 16) & 1u);   // round-to-nearest-even
  return (unsigned short)(r >> 16);
}

// monotonic float -> uint encoding for atomicMax on floats
__device__ __forceinline__ unsigned fenc(float f) {
  unsigned u = __float_as_uint(f);
  return (u & 0x80000000u) ? ~u : (u | 0x80000000u);
}
__device__ __forceinline__ float fdec(unsigned u) {
  unsigned b = (u & 0x80000000u) ? (u & 0x7FFFFFFFu) : ~u;
  return __uint_as_float(b);
}

__device__ __forceinline__ float wred(float v) {
#pragma unroll
  for (int o = 16; o > 0; o >>= 1) v += __shfl_xor(v, o, 32);
  return v;
}

__device__ __forceinline__ void edge_sd(const int* ei, int E, int t, int& s, int& d) {
  if (t < E) { s = ei[t]; d = ei[E + t]; } else { s = t - E; d = t - E; }  // self loops
}

// ------------------------------ fills --------------------------------------
__global__ void k_fill_f32(float* p, float v, long n) {
  long i = (long)blockIdx.x * blockDim.x + threadIdx.x;
  if (i < n) p[i] = v;
}
__global__ void k_fill_u32(unsigned* p, unsigned v, long n) {
  long i = (long)blockIdx.x * blockDim.x + threadIdx.x;
  if (i < n) p[i] = v;
}
__global__ void k_bias_init(float* out, const float* __restrict__ b, long n, int mask) {
  long i = (long)blockIdx.x * blockDim.x + threadIdx.x;
  if (i < n) out[i] = b[i & mask];
}

// -------------------- cosine relevance + scale + bf16 cast ------------------
__global__ void k_cos_scale(const float* __restrict__ x, const float* __restrict__ claim,
                            const int* __restrict__ batch, unsigned short* __restrict__ h0,
                            int N) {
  const int DIN = 768;
  int wid  = threadIdx.x >> 5;
  int lane = threadIdx.x & 31;
  int node = blockIdx.x * (blockDim.x >> 5) + wid;
  if (node >= N) return;
  const float* xr = x + (size_t)node * DIN;
  const float* cr = claim + (size_t)batch[node] * DIN;
  float dot = 0.f, nx = 0.f, nc = 0.f;
  float xv[24];
#pragma unroll
  for (int i = 0; i < 24; ++i) {
    int k = lane + (i << 5);
    float xe = xr[k], ce = cr[k];
    xv[i] = xe;
    dot += xe * ce; nx += xe * xe; nc += ce * ce;
  }
  dot = wred(dot); nx = wred(nx); nc = wred(nc);
  float rel = dot / fmaxf(sqrtf(nx) * sqrtf(nc), 1e-8f);
  unsigned short* hr = h0 + (size_t)node * DIN;
#pragma unroll
  for (int i = 0; i < 24; ++i) hr[lane + (i << 5)] = f2bf(xv[i] * rel);
}

// --------------- pack weight matrix into WMMA B-fragment layout -------------
// B 32x16 bf16 fragment (ISA 7.12.2): lane l holds column (l&15);
// lanes 0-15: K {0..7, 16..23}; lanes 16-31: K {8..15, 24..31}.
__global__ void k_pack_w(const float* __restrict__ W, unsigned short* __restrict__ Wp,
                         int K, int Nc) {
  int t = blockIdx.x * blockDim.x + threadIdx.x;
  int ntiles = Nc >> 4;
  int total = (K >> 5) * ntiles * 32;
  if (t >= total) return;
  int lane = t & 31;
  int rest = t >> 5;
  int nt = rest % ntiles;
  int kt = rest / ntiles;
  int col = (nt << 4) + (lane & 15);
  int kb  = (kt << 5) + ((lane & 16) ? 8 : 0);
  unsigned short* dst = Wp + (size_t)t * 16;
#pragma unroll
  for (int j = 0; j < 8; ++j) {
    dst[j]     = f2bf(W[(size_t)(kb + j) * Nc + col]);
    dst[8 + j] = f2bf(W[(size_t)(kb + 16 + j) * Nc + col]);
  }
}

// --------------------------- bf16 WMMA GEMM --------------------------------
// C[M,Nc] = A[M,K](bf16 row-major) * W(pre-packed fragments), f32 accumulate.
// 8 waves / block; each wave computes a 16-row x 64-col tile.
__global__ void k_gemm_bf16(const unsigned short* __restrict__ A,
                            const unsigned short* __restrict__ Wp,
                            float* __restrict__ C, int M, int K, int Nc) {
  int wid  = threadIdx.x >> 5;
  int lane = threadIdx.x & 31;
  int rowTile = blockIdx.x * (blockDim.x >> 5) + wid;
  if (rowTile >= (M >> 4)) return;
  int ntiles = Nc >> 4;
  int l15 = lane & 15;
  int hi  = (lane >> 4) & 1;
  const unsigned short* arow = A + (size_t)(rowTile * 16 + l15) * K + (hi ? 8 : 0);
  int colBase = blockIdx.y * 64;
  int nt0 = colBase >> 4;
  v8f acc0 = {}, acc1 = {}, acc2 = {}, acc3 = {};
  int ktn = K >> 5;
  for (int kt = 0; kt < ktn; ++kt) {
    if (kt + 1 < ktn) __builtin_prefetch(arow + (kt + 1) * 32, 0, 1);
    v8bf alo = *(const v8bf*)(arow + kt * 32);
    v8bf ahi = *(const v8bf*)(arow + kt * 32 + 16);
    v16bf a = __builtin_shufflevector(alo, ahi, 0,1,2,3,4,5,6,7,8,9,10,11,12,13,14,15);

    const unsigned short* bb = Wp + ((size_t)(kt * ntiles + nt0) * 32 + lane) * 16;
    v8bf b0l = *(const v8bf*)(bb +    0), b0h = *(const v8bf*)(bb +    8);
    v8bf b1l = *(const v8bf*)(bb +  512), b1h = *(const v8bf*)(bb +  520);
    v8bf b2l = *(const v8bf*)(bb + 1024), b2h = *(const v8bf*)(bb + 1032);
    v8bf b3l = *(const v8bf*)(bb + 1536), b3h = *(const v8bf*)(bb + 1544);
    v16bf b0 = __builtin_shufflevector(b0l, b0h, 0,1,2,3,4,5,6,7,8,9,10,11,12,13,14,15);
    v16bf b1 = __builtin_shufflevector(b1l, b1h, 0,1,2,3,4,5,6,7,8,9,10,11,12,13,14,15);
    v16bf b2 = __builtin_shufflevector(b2l, b2h, 0,1,2,3,4,5,6,7,8,9,10,11,12,13,14,15);
    v16bf b3 = __builtin_shufflevector(b3l, b3h, 0,1,2,3,4,5,6,7,8,9,10,11,12,13,14,15);

    acc0 = __builtin_amdgcn_wmma_f32_16x16x32_bf16(false, a, false, b0, (short)0, acc0, false, false);
    acc1 = __builtin_amdgcn_wmma_f32_16x16x32_bf16(false, a, false, b1, (short)0, acc1, false, false);
    acc2 = __builtin_amdgcn_wmma_f32_16x16x32_bf16(false, a, false, b2, (short)0, acc2, false, false);
    acc3 = __builtin_amdgcn_wmma_f32_16x16x32_bf16(false, a, false, b3, (short)0, acc3, false, false);
  }
  // C/D layout: lanes 0-15: VGPR r -> row r; lanes 16-31: VGPR r -> row r+8; col = lane&15
  int row0 = rowTile * 16 + hi * 8;
  float* crow = C + (size_t)row0 * Nc + colBase + l15;
#pragma unroll
  for (int r = 0; r < 8; ++r) {
    crow[(size_t)r * Nc +  0] = acc0[r];
    crow[(size_t)r * Nc + 16] = acc1[r];
    crow[(size_t)r * Nc + 32] = acc2[r];
    crow[(size_t)r * Nc + 48] = acc3[r];
  }
}

// -------------------------- attention alphas --------------------------------
__global__ void k_alpha(const float* __restrict__ H, const float* __restrict__ a_s,
                        const float* __restrict__ a_d, float* __restrict__ as_o,
                        float* __restrict__ ad_o, int N, int D) {
  int wid = threadIdx.x >> 5, lane = threadIdx.x & 31;
  int node = blockIdx.x * (blockDim.x >> 5) + wid;
  if (node >= N) return;
  const float* hr = H + (size_t)node * D;
  float s = 0.f, d = 0.f;
  for (int k = lane; k < D; k += 32) { float v = hr[k]; s += v * a_s[k]; d += v * a_d[k]; }
  s = wred(s); d = wred(d);
  if (lane == 0) { as_o[node] = s; ad_o[node] = d; }
}

// ------------------------- segment softmax passes ---------------------------
__global__ void k_edge_max(const int* __restrict__ ei, int E, int ET,
                           const float* __restrict__ as_, const float* __restrict__ ad_,
                           float* __restrict__ ebuf, unsigned* __restrict__ menc) {
  int t = blockIdx.x * blockDim.x + threadIdx.x;
  if (t >= ET) return;
  int s, d; edge_sd(ei, E, t, s, d);
  float e = as_[s] + ad_[d];
  e = (e > 0.f) ? e : 0.2f * e;          // leaky relu, PyG default slope
  ebuf[t] = e;
  atomicMax(&menc[d], fenc(e));
}

__global__ void k_edge_exp(const int* __restrict__ ei, int E, int ET,
                           float* __restrict__ ebuf, const unsigned* __restrict__ menc,
                           float* __restrict__ denom) {
  int t = blockIdx.x * blockDim.x + threadIdx.x;
  if (t >= ET) return;
  int s, d; edge_sd(ei, E, t, s, d);
  float ex = __expf(ebuf[t] - fdec(menc[d]));
  ebuf[t] = ex;
  atomicAdd(&denom[d], ex);
}

// one block (256 thr) per edge: out[dst,c] += attn * H[src,c]
__global__ void k_aggregate(const int* __restrict__ ei, int E,
                            const float* __restrict__ H, const float* __restrict__ ebuf,
                            const float* __restrict__ denom, float* __restrict__ out, int D) {
  int t = blockIdx.x;
  int c = threadIdx.x;
  int s, d; edge_sd(ei, E, t, s, d);
  float attn = ebuf[t] / denom[d];
  atomicAdd(&out[(size_t)d * D + c], H[(size_t)s * D + c] * attn);
}

// ------------------------------ batchnorm -----------------------------------
__global__ void k_bn_stats(const float* __restrict__ H, float* sum, float* sq, int N, int D) {
  int c = threadIdx.x;                      // blockDim == D
  float ls = 0.f, lq = 0.f;
  for (int n = blockIdx.x; n < N; n += gridDim.x) {
    float v = H[(size_t)n * D + c];
    ls += v; lq += v * v;
  }
  atomicAdd(&sum[c], ls);
  atomicAdd(&sq[c], lq);
}

__global__ void k_bn_apply(float* __restrict__ H, unsigned short* __restrict__ Hbf,
                           const float* __restrict__ sum, const float* __restrict__ sq,
                           const float* __restrict__ gamma, const float* __restrict__ beta,
                           int N, int D) {
  long i = (long)blockIdx.x * blockDim.x + threadIdx.x;
  if (i >= (long)N * D) return;
  int c = (int)(i & (D - 1));
  float invN = 1.0f / (float)N;
  float mu = sum[c] * invN;
  float var = sq[c] * invN - mu * mu;
  float v = (H[i] - mu) * rsqrtf(var + 1e-5f) * gamma[c] + beta[c];
  v = fmaxf(v, 0.f);                        // ReLU
  H[i]   = v;
  Hbf[i] = f2bf(v);
}

// ------------------------------- pooling ------------------------------------
__global__ void k_counts(const int* __restrict__ batch, float* counts, int N) {
  int t = blockIdx.x * blockDim.x + threadIdx.x;
  if (t < N) atomicAdd(&counts[batch[t]], 1.f);
}
__global__ void k_pool(const float* __restrict__ H, const int* __restrict__ batch,
                       float* __restrict__ pooled, int N, int D) {
  long i = (long)blockIdx.x * blockDim.x + threadIdx.x;
  if (i >= (long)N * D) return;
  int n = (int)(i >> 8);
  int c = (int)(i & 255);
  float v = fmaxf(H[i], 0.f);               // final ReLU fused into pooling
  atomicAdd(&pooled[(size_t)batch[n] * D + c], v);
}

// ------------------------------ classifier ----------------------------------
__global__ void k_clf(const float* __restrict__ pooled, const float* __restrict__ counts,
                      const float* __restrict__ claim, const float* __restrict__ W,
                      const float* __restrict__ b, float* __restrict__ out,
                      int G, int D, int DIN) {
  int wid = threadIdx.x >> 5, lane = threadIdx.x & 31;
  int g = blockIdx.x * (blockDim.x >> 5) + wid;
  if (g >= G) return;
  float den = fmaxf(counts[g], 1.f);
  float acc = 0.f;
  for (int c = lane; c < D; c += 32)   acc += (pooled[(size_t)g * D + c] / den) * W[c];
  for (int k = lane; k < DIN; k += 32) acc += claim[(size_t)g * DIN + k] * W[D + k];
  acc = wred(acc);
  if (lane == 0) out[g] = acc + b[0];
}

// ===========================================================================
extern "C" void kernel_launch(void* const* d_in, const int* in_sizes, int n_in,
                              void* d_out, int out_size, void* d_ws, size_t ws_size,
                              hipStream_t stream) {
  const float* claim = (const float*)d_in[0];
  const float* x     = (const float*)d_in[1];
  const int*   ei    = (const int*)d_in[2];
  const int*   batch = (const int*)d_in[3];
  const float* W1    = (const float*)d_in[4];
  const float* as1v  = (const float*)d_in[5];
  const float* ad1v  = (const float*)d_in[6];
  const float* b1    = (const float*)d_in[7];
  const float* W2    = (const float*)d_in[8];
  const float* as2v  = (const float*)d_in[9];
  const float* ad2v  = (const float*)d_in[10];
  const float* b2    = (const float*)d_in[11];
  const float* gamma = (const float*)d_in[12];
  const float* beta  = (const float*)d_in[13];
  const float* clfW  = (const float*)d_in[14];
  const float* clfb  = (const float*)d_in[15];
  float* out = (float*)d_out;

  const int DIN = 768, D = 256;
  const int G  = in_sizes[0] / DIN;   // 128
  const int N  = in_sizes[1] / DIN;   // 50000 (multiple of 16)
  const int E  = in_sizes[2] / 2;     // 400000
  const int ET = E + N;               // + self loops

  char* ws = (char*)d_ws;
  size_t off = 0;
  auto alloc = [&](size_t bytes) {
    size_t p = off;
    off += (bytes + 255) & ~(size_t)255;
    return p;
  };
  size_t o_h0   = alloc((size_t)N * DIN * 2);                       // bf16 scaled x
  size_t o_w1p  = alloc((size_t)(DIN / 32) * (D / 16) * 512 * 2);   // packed W1 frags
  size_t o_w2p  = alloc((size_t)(D / 32) * (D / 16) * 512 * 2);     // packed W2 frags
  size_t o_h1   = alloc((size_t)N * D * 4);                         // GAT1 linear out
  size_t o_as   = alloc((size_t)N * 4);
  size_t o_ad   = alloc((size_t)N * 4);
  size_t o_menc = alloc((size_t)N * 4);
  size_t o_den  = alloc((size_t)N * 4);
  size_t o_e    = alloc((size_t)ET * 4);
  size_t o_out1 = alloc((size_t)N * D * 4);                         // GAT1 agg / BN in place
  size_t o_hbf  = alloc((size_t)N * D * 2);                         // bf16 post-BN
  size_t o_sum  = alloc(D * 4);
  size_t o_sq   = alloc(D * 4);
  size_t o_pool = alloc((size_t)G * D * 4);
  size_t o_cnt  = alloc((size_t)G * 4);
  size_t o_h2   = o_h0;   // reuse: h0 dead after GEMM1 (N*D*4 <= N*DIN*2)
  size_t o_out2 = o_h1;   // reuse: h1 dead after layer-1 aggregation

  unsigned short* h0   = (unsigned short*)(ws + o_h0);
  unsigned short* w1p  = (unsigned short*)(ws + o_w1p);
  unsigned short* w2p  = (unsigned short*)(ws + o_w2p);
  float* h1            = (float*)(ws + o_h1);
  float* as_           = (float*)(ws + o_as);
  float* ad_           = (float*)(ws + o_ad);
  unsigned* menc       = (unsigned*)(ws + o_menc);
  float* den           = (float*)(ws + o_den);
  float* ebuf          = (float*)(ws + o_e);
  float* out1          = (float*)(ws + o_out1);
  unsigned short* hbf  = (unsigned short*)(ws + o_hbf);
  float* bnsum         = (float*)(ws + o_sum);
  float* bnsq          = (float*)(ws + o_sq);
  float* pooled        = (float*)(ws + o_pool);
  float* counts        = (float*)(ws + o_cnt);
  float* h2            = (float*)(ws + o_h2);
  float* out2          = (float*)(ws + o_out2);

  const int TB = 256;
  const long nd = (long)N * D;
  auto g1d = [](long n) { return (int)((n + 255) / 256); };

  // ---- prep: relevance scaling + weight packing ----
  k_cos_scale<<<(N + 7) / 8, TB, 0, stream>>>(x, claim, batch, h0, N);
  k_pack_w<<<g1d((long)(DIN / 32) * (D / 16) * 32), TB, 0, stream>>>(W1, w1p, DIN, D);
  k_pack_w<<<g1d((long)(D / 32) * (D / 16) * 32), TB, 0, stream>>>(W2, w2p, D, D);

  // ---- layer 1: GEMM (WMMA bf16) ----
  dim3 gg((N / 16 + 7) / 8, D / 64);
  k_gemm_bf16<<<gg, TB, 0, stream>>>(h0, w1p, h1, N, DIN, D);

  // attention (segment softmax over dst)
  k_alpha<<<(N + 7) / 8, TB, 0, stream>>>(h1, as1v, ad1v, as_, ad_, N, D);
  k_fill_u32<<<g1d(N), TB, 0, stream>>>(menc, 0u, N);
  k_fill_f32<<<g1d(N), TB, 0, stream>>>(den, 0.f, N);
  k_edge_max<<<g1d(ET), TB, 0, stream>>>(ei, E, ET, as_, ad_, ebuf, menc);
  k_edge_exp<<<g1d(ET), TB, 0, stream>>>(ei, E, ET, ebuf, menc, den);
  k_bias_init<<<g1d(nd), TB, 0, stream>>>(out1, b1, nd, D - 1);
  k_aggregate<<<ET, D, 0, stream>>>(ei, E, h1, ebuf, den, out1, D);

  // BN (training-mode batch stats) + ReLU + bf16 recast
  k_fill_f32<<<1, TB, 0, stream>>>(bnsum, 0.f, D);
  k_fill_f32<<<1, TB, 0, stream>>>(bnsq, 0.f, D);
  k_bn_stats<<<256, D, 0, stream>>>(out1, bnsum, bnsq, N, D);
  k_bn_apply<<<g1d(nd), TB, 0, stream>>>(out1, hbf, bnsum, bnsq, gamma, beta, N, D);

  // ---- layer 2: GEMM (WMMA bf16) ----
  k_gemm_bf16<<<gg, TB, 0, stream>>>(hbf, w2p, h2, N, D, D);

  k_alpha<<<(N + 7) / 8, TB, 0, stream>>>(h2, as2v, ad2v, as_, ad_, N, D);
  k_fill_u32<<<g1d(N), TB, 0, stream>>>(menc, 0u, N);
  k_fill_f32<<<g1d(N), TB, 0, stream>>>(den, 0.f, N);
  k_edge_max<<<g1d(ET), TB, 0, stream>>>(ei, E, ET, as_, ad_, ebuf, menc);
  k_edge_exp<<<g1d(ET), TB, 0, stream>>>(ei, E, ET, ebuf, menc, den);
  k_bias_init<<<g1d(nd), TB, 0, stream>>>(out2, b2, nd, D - 1);
  k_aggregate<<<ET, D, 0, stream>>>(ei, E, h2, ebuf, den, out2, D);

  // ---- ReLU + mean pool + classifier ----
  k_fill_f32<<<g1d((long)G * D), TB, 0, stream>>>(pooled, 0.f, (long)G * D);
  k_fill_f32<<<1, TB, 0, stream>>>(counts, 0.f, G);
  k_counts<<<g1d(N), TB, 0, stream>>>(batch, counts, N);
  k_pool<<<g1d(nd), TB, 0, stream>>>(out2, batch, pooled, N, D);
  k_clf<<<(G + 7) / 8, TB, 0, stream>>>(pooled, counts, claim, clfW, clfb, out, G, D, DIN);
}
